// SAGATEmbedding_575525618147
// MI455X (gfx1250) — compile-verified, matrix operand order checked by reading, not run on hardware
//
#include <hip/hip_runtime.h>
#include <math.h>

// ---------------------------------------------------------------------------
// SAGATEmbedding for MI455X (gfx1250, wave32, WMMA).
//
//   build_h0              : h0 = concat(x, joint_emb[type]) -> bf16 [N x 32]
//   per layer l=0..2:
//     pack_wcat           : [Wl_std|Wr_std|Wl_rep|Wr_rep] -> bf16 packed into
//                           the WMMA B-operand per-lane register layout
//     wmma_node_linear<K> : xlr = h @ Wcat + bias  (v_wmma_f32_16x16x32_bf16,
//                           vector b128 operand loads, K-loop fully unrolled)
//     attention_kernel    : fused edge-MLP + GATv2 softmax over the 17 known
//                           in-edges per node + rep branch + LayerNorm + ELU
//   pack_proj / proj_wmma : emb = h @ proj_W + proj_b  (WMMA, K=32, N=64)
//   final_ln_norm         : LayerNorm + L2 normalize -> d_out [N x 64] fp32
//
// Graph structure is analytic: node n = p*17 + j has in-edges from the 16
// other joints of person p and from node (p^1)*17 + j (cross-person edge,
// same joint type => the single masked-in edge of the "rep" GAT, alpha==1).
//
// Input pointer order (jax pytree flatten, dict keys sorted):
//   0:x 1:edge_index 2:joint_types 3:fln_b 4:fln_g 5:joint_emb
//   6+20*l: eW1 eW2 eb1 eb2 ln_b ln_g
//           rep{We Wl Wr att bias bl br} std{We Wl Wr att bias bl br}
//   66:proj_W 67:proj_b
// ---------------------------------------------------------------------------

#define J 17
#define PPL 1024
#define NN (PPL * J)   // 17408

typedef __attribute__((ext_vector_type(16))) __bf16 v16bf;
typedef __attribute__((ext_vector_type(8)))  __bf16 v8bf;
typedef __attribute__((ext_vector_type(8)))  float  v8f;

// ---- compile-time skeleton tables (BFS hops, edge category) ---------------
struct Tables { float hops[J][J]; int cat[J][J]; };

constexpr Tables make_tables() {
  Tables t{};
  int ea[16] = {0,0,1,2,5,7,6,8,5,5,6,11,11,13,12,14};
  int eb[16] = {1,2,3,4,7,9,8,10,6,11,12,12,13,15,14,16};
  bool adj[J][J] = {};
  for (int i = 0; i < 16; i++) { adj[ea[i]][eb[i]] = true; adj[eb[i]][ea[i]] = true; }
  for (int s = 0; s < J; s++) {
    int dist[J]; for (int i = 0; i < J; i++) dist[i] = -1;
    int q[J]; int qh = 0, qt = 0; dist[s] = 0; q[qt++] = s;
    while (qh < qt) {
      int u = q[qh++];
      for (int v = 0; v < J; v++)
        if (adj[u][v] && dist[v] < 0) { dist[v] = dist[u] + 1; q[qt++] = v; }
    }
    for (int d = 0; d < J; d++) t.hops[s][d] = (dist[d] >= 0) ? (float)dist[d] : 8.0f;
  }
  int LS[6][5] = {{0,1,2,3,4},{5,7,9,0,0},{6,8,10,0,0},{5,6,11,12,0},{11,13,15,0,0},{12,14,16,0,0}};
  int LN[6] = {5,3,3,4,3,3};
  bool limb[J][J] = {};
  for (int L = 0; L < 6; L++)
    for (int i = 0; i < LN[L]; i++)
      for (int k = 0; k < LN[L]; k++) {
        int a = LS[L][i], b = LS[L][k];
        if (a != b && !adj[a][b]) limb[a][b] = true;
      }
  for (int a = 0; a < J; a++)
    for (int b = 0; b < J; b++)
      t.cat[a][b] = (a == b) ? 0 : (adj[a][b] ? 1 : (limb[a][b] ? 2 : 3));
  return t;
}
__constant__ Tables g_tab = make_tables();

// ---- helpers --------------------------------------------------------------
__device__ inline __bf16 to_bf16(float f) {
  unsigned int u = __builtin_bit_cast(unsigned int, f);
  u = u + 0x7fffu + ((u >> 16) & 1u);       // round-to-nearest-even
  unsigned short h = (unsigned short)(u >> 16);
  return __builtin_bit_cast(__bf16, h);
}

__device__ inline float wred(float v) {
#pragma unroll
  for (int o = 16; o > 0; o >>= 1) v += __shfl_xor(v, o, 32);
  return v;
}

__device__ inline float eluf(float v) { return v > 0.f ? v : (__expf(v) - 1.f); }

// A tile: 16 rows x 32 K (bf16), row-major with `stride` elements per row.
// ISA layout: lane<16 -> row=lane, K kb..kb+7 & kb+16..kb+23 with kb=0;
// lane>=16 -> same rows, kb=8. Two contiguous 16-byte vector loads per lane.
__device__ inline v16bf load_a_tile(const __bf16* base, int stride, int row0, int k0, int lane) {
  int row = row0 + (lane & 15);
  int kb = k0 + ((lane >> 4) << 3);
  const __bf16* p = base + row * stride + kb;
  v8bf lo = *(const v8bf*)p;          // K = kb .. kb+7
  v8bf hi = *(const v8bf*)(p + 16);   // K = kb+16 .. kb+23
  return __builtin_shufflevector(lo, hi, 0,1,2,3,4,5,6,7,8,9,10,11,12,13,14,15);
}

// B operand pre-packed in per-lane register layout:
//   bpk[(((ct*KC)+kc)*32 + lane)*16 + j]  -> one 32-byte contiguous load.
__device__ inline v16bf load_b_packed(const __bf16* bpk, int KC, int ct, int kc, int lane) {
  return *(const v16bf*)(bpk + (size_t)(((ct * KC) + kc) * 32 + lane) * 16);
}

// ---- kernels --------------------------------------------------------------
__global__ __launch_bounds__(256)
void build_h0(const float* __restrict__ x, const int* __restrict__ jt,
              const float* __restrict__ jemb, __bf16* __restrict__ hb) {
  int g = blockIdx.x * 256 + threadIdx.x;     // NN*32 threads
  int n = g >> 5, c = g & 31;
  float v = (c < 16) ? x[n * 16 + c] : jemb[jt[n] * 16 + (c - 16)];
  hb[n * 128 + c] = to_bf16(v);
}

// Pack [Wl_std|Wr_std|Wl_rep|Wr_rep] (logical [ind x 256]) directly into the
// WMMA B-operand register layout; also emit the 256-wide bias vector.
__global__ __launch_bounds__(256)
void pack_wcat(const float* __restrict__ WlS, const float* __restrict__ WrS,
               const float* __restrict__ WlR, const float* __restrict__ WrR,
               const float* __restrict__ blS, const float* __restrict__ brS,
               const float* __restrict__ blR, const float* __restrict__ brR,
               __bf16* __restrict__ bpk, float* __restrict__ bias, int ind) {
  int KC = ind >> 5;
  int gid = blockIdx.x * 256 + threadIdx.x;   // ind*256 elements
  if (gid < ind * 256) {
    int jj   = gid & 15;
    int lane = (gid >> 4) & 31;
    int rem  = gid >> 9;
    int kc   = rem % KC;
    int ct   = rem / KC;
    int col  = ct * 16 + (lane & 15);
    int kk   = kc * 32 + ((lane >> 4) << 3) + ((jj < 8) ? jj : (8 + jj));
    float v;
    if (col < 96)       v = WlS[kk * 96 + col];
    else if (col < 192) v = WrS[kk * 96 + (col - 96)];
    else if (col < 224) v = WlR[kk * 32 + (col - 192)];
    else                v = WrR[kk * 32 + (col - 224)];
    bpk[gid] = to_bf16(v);
  }
  if (gid < 256) {
    int c = gid; float v;
    if (c < 96)       v = blS[c];
    else if (c < 192) v = brS[c - 96];
    else if (c < 224) v = blR[c - 192];
    else              v = brR[c - 224];
    bias[c] = v;
  }
}

// Pack proj_W [32 x 64] into B-operand layout (KC=1, 4 col tiles).
__global__ __launch_bounds__(256)
void pack_proj(const float* __restrict__ pw, __bf16* __restrict__ ppk) {
  int gid = blockIdx.x * 256 + threadIdx.x;   // 2048 elements
  if (gid < 2048) {
    int jj   = gid & 15;
    int lane = (gid >> 4) & 31;
    int ct   = gid >> 9;
    int col  = ct * 16 + (lane & 15);
    int kk   = ((lane >> 4) << 3) + ((jj < 8) ? jj : (8 + jj));
    ppk[gid] = to_bf16(pw[kk * 64 + col]);
  }
}

// xlr[n][256] = [xl_std(96) | xr_std(96) | xl_rep(32) | xr_rep(32)]
template <int IND>
__global__ __launch_bounds__(256)
void wmma_node_linear(const __bf16* __restrict__ hb, const __bf16* __restrict__ bpk,
                      const float* __restrict__ bias, float* __restrict__ xlr) {
  constexpr int KC = IND / 32;
  int wave = threadIdx.x >> 5, lane = threadIdx.x & 31;
  int tile = blockIdx.x * 8 + wave;           // (NN/16)*16 tiles
  int rt = tile >> 4, ct = tile & 15;
  int row0 = rt * 16, col0 = ct * 16;
  float bv = bias[col0 + (lane & 15)];
  v8f c;
#pragma unroll
  for (int r = 0; r < 8; r++) c[r] = bv;
#pragma unroll
  for (int kc = 0; kc < KC; kc++) {
    v16bf a = load_a_tile(hb, 128, row0, kc * 32, lane);
    v16bf b = load_b_packed(bpk, KC, ct, kc, lane);
    c = __builtin_amdgcn_wmma_f32_16x16x32_bf16(false, a, false, b, (short)0, c, false, false);
  }
  int mb = (lane >> 4) << 3;
  int col = col0 + (lane & 15);
#pragma unroll
  for (int r = 0; r < 8; r++) xlr[(row0 + mb + r) * 256 + col] = c[r];
}

// Fused edge MLP + GATv2 attention + rep branch + LN + ELU. One wave per node.
__global__ __launch_bounds__(256)
void attention_kernel(const float* __restrict__ xlr, const float* __restrict__ x,
                      const float* __restrict__ eW1, const float* __restrict__ eb1,
                      const float* __restrict__ eW2, const float* __restrict__ eb2,
                      const float* __restrict__ WeS, const float* __restrict__ attS,
                      const float* __restrict__ biasS, const float* __restrict__ biasR,
                      const float* __restrict__ lng, const float* __restrict__ lnb,
                      __bf16* __restrict__ hb, int last) {
  __shared__ float s_eW1[224], s_eb1[32], s_eW2[512], s_eb2[16];
  __shared__ float s_We[1536], s_att[96], s_bS[96], s_bR[32];
  __shared__ float s_g[128], s_b[128];
  __shared__ float s_hid[8][32], s_ea[8][16], s_lg[8][J * 3];

  int tid = threadIdx.x;
  if (tid < 224) s_eW1[tid] = eW1[tid];
  if (tid < 32)  s_eb1[tid] = eb1[tid];
  for (int i = tid; i < 512; i += 256) s_eW2[i] = eW2[i];
  if (tid < 16)  s_eb2[tid] = eb2[tid];
  for (int i = tid; i < 1536; i += 256) s_We[i] = WeS[i];
  if (tid < 96) { s_att[tid] = attS[tid]; s_bS[tid] = biasS[tid]; }
  if (tid < 32)  s_bR[tid] = biasR[tid];
  int lnd = last ? 32 : 128;
  if (tid < lnd) { s_g[tid] = lng[tid]; s_b[tid] = lnb[tid]; }
  __syncthreads();

  int wv = tid >> 5, d = tid & 31;
  int n = blockIdx.x * 8 + wv;
  int p = n / J, j = n - p * J;
  int partner = (p ^ 1) * J + j;
  float pnx = x[n * 16], pny = x[n * 16 + 1];
  const float* xn = xlr + (long)n * 256;
  float xr0 = xn[96 + d], xr1 = xn[128 + d], xr2 = xn[160 + d];

  // pass 1: logits for the 3 std heads over the 17 in-edges
  for (int k = 0; k < J; k++) {
    int src; float samef, hopv; int cat;
    if (k == 16) { src = partner; samef = 1.f; cat = 0; hopv = 0.f; }
    else {
      int sj = k + (k >= j);
      src = p * J + sj; samef = 0.f;
      cat = g_tab.cat[sj][j]; hopv = g_tab.hops[sj][j];
    }
    float dx = x[src * 16] - pnx, dy = x[src * 16 + 1] - pny;
    float sd = sqrtf(dx * dx + dy * dy);
    float ef0 = (cat == 0), ef1 = (cat == 1), ef2 = (cat == 2), ef3 = (cat == 3);
    float ef4 = sd, ef5 = hopv * 0.125f, ef6 = samef;

    float hid = s_eb1[d]
              + ef0 * s_eW1[d]        + ef1 * s_eW1[32 + d]
              + ef2 * s_eW1[64 + d]   + ef3 * s_eW1[96 + d]
              + ef4 * s_eW1[128 + d]  + ef5 * s_eW1[160 + d]
              + ef6 * s_eW1[192 + d];
    hid = fmaxf(hid, 0.f);
    s_hid[wv][d] = hid;
    __syncthreads();
    if (d < 16) {
      float s = s_eb2[d];
#pragma unroll
      for (int t = 0; t < 32; t++) s += s_hid[wv][t] * s_eW2[t * 16 + d];
      s_ea[wv][d] = s;
    }
    __syncthreads();

    float e0 = 0.f, e1 = 0.f, e2 = 0.f;
#pragma unroll
    for (int q = 0; q < 16; q++) {
      float a = s_ea[wv][q];
      e0 += a * s_We[q * 96 + d];
      e1 += a * s_We[q * 96 + 32 + d];
      e2 += a * s_We[q * 96 + 64 + d];
    }
    const float* xs = xlr + (long)src * 256;
    float m0 = xs[d]      + xr0 + e0;
    float m1 = xs[32 + d] + xr1 + e1;
    float m2 = xs[64 + d] + xr2 + e2;
    m0 = (m0 > 0.f) ? m0 : 0.2f * m0;
    m1 = (m1 > 0.f) ? m1 : 0.2f * m1;
    m2 = (m2 > 0.f) ? m2 : 0.2f * m2;
    float c0 = wred(m0 * s_att[d]);
    float c1 = wred(m1 * s_att[32 + d]);
    float c2 = wred(m2 * s_att[64 + d]);
    if (d == 0) { s_lg[wv][k * 3] = c0; s_lg[wv][k * 3 + 1] = c1; s_lg[wv][k * 3 + 2] = c2; }
    __syncthreads();
  }

  // softmax parameters (all lanes, redundant — 51 LDS reads)
  float mx0 = -1e30f, mx1 = -1e30f, mx2 = -1e30f;
  for (int k = 0; k < J; k++) {
    mx0 = fmaxf(mx0, s_lg[wv][k * 3]);
    mx1 = fmaxf(mx1, s_lg[wv][k * 3 + 1]);
    mx2 = fmaxf(mx2, s_lg[wv][k * 3 + 2]);
  }
  float s0 = 0.f, s1 = 0.f, s2 = 0.f;
  for (int k = 0; k < J; k++) {
    s0 += __expf(s_lg[wv][k * 3]     - mx0);
    s1 += __expf(s_lg[wv][k * 3 + 1] - mx1);
    s2 += __expf(s_lg[wv][k * 3 + 2] - mx2);
  }
  float i0 = 1.f / (s0 + 1e-16f), i1 = 1.f / (s1 + 1e-16f), i2 = 1.f / (s2 + 1e-16f);

  // pass 2: weighted aggregation of xl_std
  float a0 = 0.f, a1 = 0.f, a2 = 0.f;
  for (int k = 0; k < J; k++) {
    int src = (k == 16) ? partner : (p * J + k + (k >= j));
    float w0 = __expf(s_lg[wv][k * 3]     - mx0) * i0;
    float w1 = __expf(s_lg[wv][k * 3 + 1] - mx1) * i1;
    float w2 = __expf(s_lg[wv][k * 3 + 2] - mx2) * i2;
    const float* xs = xlr + (long)src * 256;
    a0 += w0 * xs[d];
    a1 += w1 * xs[32 + d];
    a2 += w2 * xs[64 + d];
  }
  float h0 = a0 + s_bS[d], h1 = a1 + s_bS[32 + d], h2 = a2 + s_bS[64 + d];
  // rep branch: single masked-in edge => softmax weight is exactly 1
  float h3 = xlr[(long)partner * 256 + 192 + d] + s_bR[d];

  if (!last) {
    float mu = wred(h0 + h1 + h2 + h3) * (1.f / 128.f);
    float var = wred(h0 * h0 + h1 * h1 + h2 * h2 + h3 * h3) * (1.f / 128.f) - mu * mu;
    float rs = rsqrtf(var + 1e-5f);
    float y0 = eluf((h0 - mu) * rs * s_g[d]      + s_b[d]);
    float y1 = eluf((h1 - mu) * rs * s_g[32 + d] + s_b[32 + d]);
    float y2 = eluf((h2 - mu) * rs * s_g[64 + d] + s_b[64 + d]);
    float y3 = eluf((h3 - mu) * rs * s_g[96 + d] + s_b[96 + d]);
    __bf16* hp = hb + (long)n * 128;
    hp[d] = to_bf16(y0); hp[32 + d] = to_bf16(y1);
    hp[64 + d] = to_bf16(y2); hp[96 + d] = to_bf16(y3);
  } else {
    float hm = (h0 + h1 + h2 + h3) * 0.25f;       // mean over 4 "heads"
    float mu = wred(hm) * (1.f / 32.f);
    float var = wred(hm * hm) * (1.f / 32.f) - mu * mu;
    float y = eluf((hm - mu) * rsqrtf(var + 1e-5f) * s_g[d] + s_b[d]);
    hb[(long)n * 128 + d] = to_bf16(y);
  }
}

__global__ __launch_bounds__(256)
void proj_wmma(const __bf16* __restrict__ hb, const __bf16* __restrict__ ppk,
               const float* __restrict__ pb, float* __restrict__ t) {
  int wave = threadIdx.x >> 5, lane = threadIdx.x & 31;
  int tile = blockIdx.x * 8 + wave;           // (NN/16)*4 tiles
  int rt = tile >> 2, ct = tile & 3;
  int row0 = rt * 16, col0 = ct * 16;
  float bv = pb[col0 + (lane & 15)];
  v8f c;
#pragma unroll
  for (int r = 0; r < 8; r++) c[r] = bv;
  v16bf a = load_a_tile(hb, 128, row0, 0, lane);
  v16bf b = load_b_packed(ppk, 1, ct, 0, lane);
  c = __builtin_amdgcn_wmma_f32_16x16x32_bf16(false, a, false, b, (short)0, c, false, false);
  int mb = (lane >> 4) << 3;
  int col = col0 + (lane & 15);
#pragma unroll
  for (int r = 0; r < 8; r++) t[(row0 + mb + r) * 64 + col] = c[r];
}

__global__ __launch_bounds__(256)
void final_ln_norm(const float* __restrict__ t, const float* __restrict__ g,
                   const float* __restrict__ b, float* __restrict__ out) {
  int wv = threadIdx.x >> 5, d = threadIdx.x & 31;
  int n = blockIdx.x * 8 + wv;
  float v0 = t[n * 64 + d], v1 = t[n * 64 + 32 + d];
  float mu = wred(v0 + v1) * (1.f / 64.f);
  float var = wred(v0 * v0 + v1 * v1) * (1.f / 64.f) - mu * mu;
  float rs = rsqrtf(var + 1e-5f);
  float y0 = (v0 - mu) * rs * g[d] + b[d];
  float y1 = (v1 - mu) * rs * g[32 + d] + b[32 + d];
  float nrm = sqrtf(wred(y0 * y0 + y1 * y1));
  float inv = 1.f / fmaxf(nrm, 1e-12f);
  out[n * 64 + d] = y0 * inv;
  out[n * 64 + 32 + d] = y1 * inv;
}

// ---------------------------------------------------------------------------
extern "C" void kernel_launch(void* const* d_in, const int* in_sizes, int n_in,
                              void* d_out, int out_size, void* d_ws, size_t ws_size,
                              hipStream_t stream) {
  (void)in_sizes; (void)n_in; (void)out_size; (void)ws_size;
  const float* x     = (const float*)d_in[0];
  const int*   jt    = (const int*)d_in[2];
  const float* fln_b = (const float*)d_in[3];
  const float* fln_g = (const float*)d_in[4];
  const float* jemb  = (const float*)d_in[5];
  const float* projW = (const float*)d_in[66];
  const float* projb = (const float*)d_in[67];

  char* ws = (char*)d_ws;
  size_t off = 0;
  auto take = [&](size_t bytes) -> char* {
    char* ptr = ws + off;
    off = (off + bytes + 255) & ~(size_t)255;
    return ptr;
  };
  __bf16* h_bf = (__bf16*)take((size_t)NN * 128 * 2);
  float*  xlr  = (float*)take((size_t)NN * 256 * 4);
  __bf16* bpk  = (__bf16*)take((size_t)128 * 256 * 2);
  float*  bcat = (float*)take(256 * 4);
  __bf16* ppk  = (__bf16*)take((size_t)32 * 64 * 2);
  float*  t64  = (float*)take((size_t)NN * 64 * 4);
  float*  out  = (float*)d_out;

  build_h0<<<(NN * 32) / 256, 256, 0, stream>>>(x, jt, jemb, h_bf);

  for (int l = 0; l < 3; l++) {
    int base = 6 + l * 20;
    const float* eW1   = (const float*)d_in[base + 0];
    const float* eW2   = (const float*)d_in[base + 1];
    const float* eb1   = (const float*)d_in[base + 2];
    const float* eb2   = (const float*)d_in[base + 3];
    const float* ln_b  = (const float*)d_in[base + 4];
    const float* ln_g  = (const float*)d_in[base + 5];
    const float* rWl   = (const float*)d_in[base + 7];
    const float* rWr   = (const float*)d_in[base + 8];
    const float* rbias = (const float*)d_in[base + 10];
    const float* rbl   = (const float*)d_in[base + 11];
    const float* rbr   = (const float*)d_in[base + 12];
    const float* sWe   = (const float*)d_in[base + 13];
    const float* sWl   = (const float*)d_in[base + 14];
    const float* sWr   = (const float*)d_in[base + 15];
    const float* satt  = (const float*)d_in[base + 16];
    const float* sbias = (const float*)d_in[base + 17];
    const float* sbl   = (const float*)d_in[base + 18];
    const float* sbr   = (const float*)d_in[base + 19];
    int ind = (l == 0) ? 32 : 128;

    pack_wcat<<<ind, 256, 0, stream>>>(sWl, sWr, rWl, rWr, sbl, sbr, rbl, rbr,
                                       bpk, bcat, ind);
    if (l == 0)
      wmma_node_linear<32><<<(NN / 16) * 16 / 8, 256, 0, stream>>>(h_bf, bpk, bcat, xlr);
    else
      wmma_node_linear<128><<<(NN / 16) * 16 / 8, 256, 0, stream>>>(h_bf, bpk, bcat, xlr);
    attention_kernel<<<NN / 8, 256, 0, stream>>>(xlr, x, eW1, eb1, eW2, eb2,
                                                 sWe, satt, sbias, rbias,
                                                 ln_g, ln_b, h_bf, (l == 2) ? 1 : 0);
  }

  pack_proj<<<8, 256, 0, stream>>>(projW, ppk);
  proj_wmma<<<(NN / 16) * 4 / 8, 256, 0, stream>>>(h_bf, ppk, projb, t64);
  final_ln_norm<<<NN / 8, 256, 0, stream>>>(t64, fln_g, fln_b, out);
}
